// LinearSelfAttentionBlock_11802570129788
// MI455X (gfx1250) — matrix operand order repaired and checked by reading, not compiled, not run on hardware
//
#include <hip/hip_runtime.h>
#include <hip/hip_bf16.h>
#include <math.h>

// Problem constants (from reference)
#define B_  4
#define L_  4096
#define D_  1024
#define H_  16
#define M_  4096
#define Dh_ 64
static constexpr float EPS_LN   = 1e-5f;
static constexpr float EPS_ATTN = 1e-6f;

typedef __attribute__((ext_vector_type(16))) __bf16 v16bf;
typedef __attribute__((ext_vector_type(8)))  float  v8f;

union FragB16 { v16bf v; unsigned int u[8]; };
union FragAcc { v8f v; float f[8]; };

__device__ __forceinline__ unsigned short f32_bf16(float f) {
  unsigned int u = __float_as_uint(f);
  u += 0x7fffu + ((u >> 16) & 1u);   // round-to-nearest-even
  return (unsigned short)(u >> 16);
}

// --------------------------- CDNA5 feature probes ---------------------------
#ifndef __has_builtin
#define __has_builtin(x) 0
#endif
#if __has_builtin(__builtin_amdgcn_tensor_load_to_lds) && \
    __has_builtin(__builtin_amdgcn_s_wait_tensorcnt)
#define HAVE_TDM 1
#else
#define HAVE_TDM 0
#endif
#if __has_builtin(__builtin_amdgcn_global_load_async_to_lds_b128) && \
    __has_builtin(__builtin_amdgcn_s_wait_asynccnt)
#define HAVE_ASYNC 1
#else
#define HAVE_ASYNC 0
#endif

#if HAVE_TDM || HAVE_ASYNC
#define AS1 __attribute__((address_space(1)))
#define AS3 __attribute__((address_space(3)))
typedef int vi4 __attribute__((ext_vector_type(4)));
#endif

#if HAVE_TDM
typedef unsigned int v4u __attribute__((ext_vector_type(4)));
typedef int          v8i __attribute__((ext_vector_type(8)));
typedef int          v4i __attribute__((ext_vector_type(4)));

// DMA a 128-row x 32-col bf16 tile (row stride ld elements) into LDS.
// D# per CDNA5 ISA: 2D tensor == tile, data_size=2B.
__device__ __forceinline__ void tdm_tile_128x32(const unsigned short* gsrc,
                                                int ld, unsigned lds_off) {
  unsigned long long ga = (unsigned long long)gsrc;
  v4u g0;
  g0.x = 1u;                                   // count=1 (valid descriptor)
  g0.y = lds_off;                              // LDS byte address
  g0.z = (unsigned)ga;                         // global addr [31:0]
  g0.w = (unsigned)((ga >> 32) & 0x1ffffffu) | (2u << 30);  // addr[56:32], type=2
  v8i g1;
  g1[0] = 0x00010000;        // wg_mask=0, data_size=1 (2 bytes)
  g1[1] = (int)(32u << 16);  // tensor_dim0 = 32 (bits 79:48, low half)
  g1[2] = (int)(128u << 16); // tensor_dim0 hi=0 | tensor_dim1 = 128 (low half)
  g1[3] = (int)(32u << 16);  // tensor_dim1 hi=0 | tile_dim0 = 32
  g1[4] = 128;               // tile_dim1 = 128, tile_dim2 = 0
  g1[5] = ld;                // tensor_dim0_stride[31:0]
  g1[6] = 0;                 // stride hi | tensor_dim1_stride lo
  g1[7] = 0;
  v4i z4 = {0, 0, 0, 0};
#if __clang_major__ >= 23
  v8i z8 = {0, 0, 0, 0, 0, 0, 0, 0};
  __builtin_amdgcn_tensor_load_to_lds(g0, g1, z4, z4, z8, 0);
#else
  __builtin_amdgcn_tensor_load_to_lds(g0, g1, z4, z4, 0);
#endif
}
__device__ __forceinline__ unsigned lds_offset_of(void* p) {
  return (unsigned)(unsigned long long)(AS3 char*)p;
}
#endif  // HAVE_TDM

#if HAVE_ASYNC
__device__ __forceinline__ void async_cp16(const unsigned short* src,
                                           unsigned short* dst) {
  __builtin_amdgcn_global_load_async_to_lds_b128(
      (AS1 vi4*)(vi4*)(void*)const_cast<unsigned short*>(src),
      (AS3 vi4*)(vi4*)(void*)dst, 0, 0);
}
// Per-lane async copy: 256 threads x 2 x 16B = one 128x32 bf16 tile.
__device__ __forceinline__ void stage_tile_async(const unsigned short* src,
                                                 int ld, unsigned short* dst,
                                                 int tid) {
#pragma unroll
  for (int i = 0; i < 2; ++i) {
    int u = tid * 2 + i, row = u >> 2, q = u & 3;
    async_cp16(src + (size_t)row * ld + 8 * q, dst + row * 32 + 8 * q);
  }
}
#endif  // HAVE_ASYNC

__device__ __forceinline__ void stage_tile_sync(const unsigned short* src,
                                                int ld, unsigned short* dst,
                                                int tid) {
#pragma unroll
  for (int i = 0; i < 2; ++i) {
    int u = tid * 2 + i, row = u >> 2, q = u & 3;
    uint4 vv = *(const uint4*)(src + (size_t)row * ld + 8 * q);
    *(uint4*)(dst + row * 32 + 8 * q) = vv;
  }
}

// ---------------------------------------------------------------------------
// f32 -> bf16 conversion (grid-stride)
// ---------------------------------------------------------------------------
__global__ void cvt_bf16_kernel(const float* __restrict__ in,
                                unsigned short* __restrict__ out, long n) {
  long i = (long)blockIdx.x * blockDim.x + threadIdx.x;
  long stride = (long)gridDim.x * blockDim.x;
  for (; i < n; i += stride) out[i] = f32_bf16(in[i]);
}

// ---------------------------------------------------------------------------
// f32 [K][N] -> bf16 [N][K] transpose (32x32 LDS tiles)
// ---------------------------------------------------------------------------
__global__ __launch_bounds__(256) void cvt_bf16_T_kernel(
    const float* __restrict__ in, unsigned short* __restrict__ out,
    int K, int N) {
  __shared__ float tile[32][33];
  const int k0 = blockIdx.y * 32, n0 = blockIdx.x * 32;
  const int tx = threadIdx.x & 31, ty = threadIdx.x >> 5;  // 8 rows / pass
#pragma unroll
  for (int i = 0; i < 32; i += 8)
    tile[ty + i][tx] = in[(size_t)(k0 + ty + i) * N + n0 + tx];
  __syncthreads();
#pragma unroll
  for (int i = 0; i < 32; i += 8)
    out[(size_t)(n0 + ty + i) * K + k0 + tx] = f32_bf16(tile[tx][ty + i]);
}

// ---------------------------------------------------------------------------
// WMMA bf16 GEMM: C[M,N] = A[M,K] @ B^T (B stored [N][K]) + bias, epilogues.
// 128x128 block tile, 8 waves x (2x4) 16x16 tiles, K-step 32, double-buffered
// LDS staged via TDM / async-to-LDS / sync fallback.
// ---------------------------------------------------------------------------
enum { EPI_NONE = 0, EPI_GELU_BF16 = 1, EPI_RES2 = 2 };

template <int EPI>
__global__ __launch_bounds__(256) void gemm_bf16_kernel(
    const unsigned short* __restrict__ A,    // [M,K] bf16 row-major
    const unsigned short* __restrict__ BmT,  // [N,K] bf16 row-major
    const float* __restrict__ bias,          // [N]
    float* __restrict__ Cf, unsigned short* __restrict__ Cb,
    const float* __restrict__ R1, const float* __restrict__ R2,
    int M, int N, int K) {
  __shared__ unsigned short As[2][128 * 32];  // [m][k]
  __shared__ unsigned short Bt[2][128 * 32];  // [n][k]
  const int tid  = threadIdx.x;
  const int lane = tid & 31;
  const int w    = tid >> 5;
  const int wm   = w & 3;
  const int wn   = w >> 2;
  const int m0   = blockIdx.y * 128;
  const int n0   = blockIdx.x * 128;
  const int l16  = lane & 15;
  const int lh   = lane >> 4;
  const unsigned short* Abase = A + (size_t)m0 * K;
  const unsigned short* Bbase = BmT + (size_t)n0 * K;

  FragAcc acc[2][4];
#pragma unroll
  for (int mi = 0; mi < 2; ++mi)
#pragma unroll
    for (int ni = 0; ni < 4; ++ni)
#pragma unroll
      for (int r = 0; r < 8; ++r) acc[mi][ni].f[r] = 0.0f;

  // ---- prologue: stage k0 = 0 into buffer 0 -------------------------------
#if HAVE_TDM
  if (tid < 32) {
    tdm_tile_128x32(Abase, K, lds_offset_of(&As[0][0]));
    tdm_tile_128x32(Bbase, K, lds_offset_of(&Bt[0][0]));
  }
#elif HAVE_ASYNC
  stage_tile_async(Abase, K, &As[0][0], tid);
  stage_tile_async(Bbase, K, &Bt[0][0], tid);
#else
  stage_tile_sync(Abase, K, &As[0][0], tid);
  stage_tile_sync(Bbase, K, &Bt[0][0], tid);
#endif

  for (int k0 = 0; k0 < K; k0 += 32) {
    const int cur = (k0 >> 5) & 1, nxt = cur ^ 1;
    const bool more = (k0 + 32) < K;
    // ---- stage next tile into the other buffer ----------------------------
    if (more) {
#if HAVE_TDM
      if (tid < 32) {
        tdm_tile_128x32(Abase + k0 + 32, K, lds_offset_of(&As[nxt][0]));
        tdm_tile_128x32(Bbase + k0 + 32, K, lds_offset_of(&Bt[nxt][0]));
      }
#elif HAVE_ASYNC
      stage_tile_async(Abase + k0 + 32, K, &As[nxt][0], tid);
      stage_tile_async(Bbase + k0 + 32, K, &Bt[nxt][0], tid);
#else
      stage_tile_sync(Abase + k0 + 32, K, &As[nxt][0], tid);
      stage_tile_sync(Bbase + k0 + 32, K, &Bt[nxt][0], tid);
#endif
    }
    // ---- wait for current buffer's DMA, then converge ---------------------
#if HAVE_TDM
    if (more) __builtin_amdgcn_s_wait_tensorcnt((short)2);
    else      __builtin_amdgcn_s_wait_tensorcnt((short)0);
#elif HAVE_ASYNC
    if (more) __builtin_amdgcn_s_wait_asynccnt((short)4);
    else      __builtin_amdgcn_s_wait_asynccnt((short)0);
#endif
    __syncthreads();

    const unsigned short* as = &As[cur][0];
    const unsigned short* bt = &Bt[cur][0];
    FragB16 bf[4];
#pragma unroll
    for (int ni = 0; ni < 4; ++ni) {
      int rowB = 64 * wn + 16 * ni + l16;
#pragma unroll
      for (int j = 0; j < 8; ++j)
        bf[ni].u[j] = *(const unsigned int*)(&bt[rowB * 32 + ((j >> 2) << 4) + (lh << 3) + ((j & 3) << 1)]);
    }
#pragma unroll
    for (int mi = 0; mi < 2; ++mi) {
      FragB16 af;
      int rowA = 32 * wm + 16 * mi + l16;
#pragma unroll
      for (int j = 0; j < 8; ++j) {
        int kk = ((j >> 2) << 4) + (lh << 3) + ((j & 3) << 1);
        af.u[j] = *(const unsigned int*)(&as[rowA * 32 + kk]);
      }
#pragma unroll
      for (int ni = 0; ni < 4; ++ni)
        acc[mi][ni].v = __builtin_amdgcn_wmma_f32_16x16x32_bf16(
            false, af.v, false, bf[ni].v, (short)0, acc[mi][ni].v, false, false);
    }
    __syncthreads();  // everyone done reading buf[cur] before it is restaged
  }

  // ---- epilogue -----------------------------------------------------------
#pragma unroll
  for (int mi = 0; mi < 2; ++mi)
#pragma unroll
    for (int ni = 0; ni < 4; ++ni)
#pragma unroll
      for (int r = 0; r < 8; ++r) {
        int m = m0 + 32 * wm + 16 * mi + 8 * lh + r;
        int n = n0 + 64 * wn + 16 * ni + l16;
        size_t idx = (size_t)m * N + n;
        float v = acc[mi][ni].f[r] + bias[n];
        if (EPI == EPI_NONE) {
          Cf[idx] = v;
        } else if (EPI == EPI_GELU_BF16) {
          float g = 0.5f * v * (1.0f + erff(v * 0.70710678118654752f));
          Cb[idx] = f32_bf16(g);
        } else {  // EPI_RES2: + a + x
          Cf[idx] = v + R1[idx] + R2[idx];
        }
      }
}

// ---------------------------------------------------------------------------
// Row LayerNorm over D_=1024; writes f32 and optionally bf16 copy.
// ---------------------------------------------------------------------------
__global__ __launch_bounds__(256) void ln_kernel(
    const float* __restrict__ X, const float* __restrict__ g,
    const float* __restrict__ b, float* __restrict__ Yf,
    unsigned short* __restrict__ Yb) {
  const int row = blockIdx.x, tid = threadIdx.x;
  __shared__ float red[256];
  const float* x = X + (size_t)row * D_;
  float vals[4];
  float s = 0.f;
#pragma unroll
  for (int i = 0; i < 4; ++i) { vals[i] = x[tid + 256 * i]; s += vals[i]; }
  red[tid] = s; __syncthreads();
  for (int st = 128; st > 0; st >>= 1) {
    if (tid < st) red[tid] += red[tid + st];
    __syncthreads();
  }
  float mean = red[0] * (1.0f / D_);
  __syncthreads();
  float sq = 0.f;
#pragma unroll
  for (int i = 0; i < 4; ++i) { float d = vals[i] - mean; sq += d * d; }
  red[tid] = sq; __syncthreads();
  for (int st = 128; st > 0; st >>= 1) {
    if (tid < st) red[tid] += red[tid + st];
    __syncthreads();
  }
  float rstd = rsqrtf(red[0] * (1.0f / D_) + EPS_LN);
#pragma unroll
  for (int i = 0; i < 4; ++i) {
    int c = tid + 256 * i;
    float y = (vals[i] - mean) * rstd * g[c] + b[c];
    if (Yf) Yf[(size_t)row * D_ + c] = y;
    if (Yb) Yb[(size_t)row * D_ + c] = f32_bf16(y);
  }
}

// ---------------------------------------------------------------------------
// phi: per (b,l,h) row of 64: t = LN(q @ Wg + bg; gg, betag);
//      phi = softplus(-t*cc)/cc + t.   4 rows per 256-thread block.
// ---------------------------------------------------------------------------
__global__ __launch_bounds__(256) void phi_kernel(
    const float* __restrict__ X,  // [B*L, D]
    const float* __restrict__ Wg, const float* __restrict__ bg,
    const float* __restrict__ gg, const float* __restrict__ betag,
    const float* __restrict__ cptr, float* __restrict__ Pf,
    unsigned short* __restrict__ Pb) {
  const int tid = threadIdx.x;
  const int grp = tid >> 6, j = tid & 63;
  const int rid = blockIdx.x * 4 + grp;   // (b*L + l)*H + h
  const int bl = rid >> 4, h = rid & 15;
  __shared__ float qs[4][64];
  __shared__ float ts[4][64];
  qs[grp][j] = X[(size_t)bl * D_ + h * 64 + j];
  __syncthreads();
  float t = bg[j];
  for (int i = 0; i < 64; ++i) t += qs[grp][i] * Wg[i * 64 + j];
  ts[grp][j] = t;
  __syncthreads();
  float s = 0.f, sq = 0.f;
  for (int i = 0; i < 64; ++i) s += ts[grp][i];
  float mean = s * (1.0f / 64.0f);
  for (int i = 0; i < 64; ++i) { float d = ts[grp][i] - mean; sq += d * d; }
  float rstd = rsqrtf(sq * (1.0f / 64.0f) + EPS_LN);
  float y = (t - mean) * rstd * gg[j] + betag[j];
  float cc = fabsf(cptr[0]) + EPS_ATTN;
  float z = -y * cc;
  float sp = (z > 20.f) ? z : ((z < -20.f) ? expf(z) : log1pf(expf(z)));
  float p = sp / cc + y;
  Pf[(size_t)rid * 64 + j] = p;
  if (Pb) Pb[(size_t)rid * 64 + j] = f32_bf16(p);
}

// ---------------------------------------------------------------------------
// q_global[b,h,d] = mean_l Qh / sqrt(Dh)  (coalesced: 4 rows in flight)
// ---------------------------------------------------------------------------
__global__ __launch_bounds__(256) void qglobal_kernel(
    const float* __restrict__ Q, float* __restrict__ qg) {
  const int bh = blockIdx.x, b = bh >> 4, h = bh & 15;
  const int d = threadIdx.x & 63, lg = threadIdx.x >> 6;
  __shared__ float ps[4][64];
  float s = 0.f;
  for (int l = lg; l < L_; l += 4)
    s += Q[(size_t)(b * L_ + l) * D_ + h * 64 + d];
  ps[lg][d] = s;
  __syncthreads();
  if (lg == 0)
    qg[bh * 64 + d] =
        (ps[0][d] + ps[1][d] + ps[2][d] + ps[3][d]) * (1.0f / (L_ * 8.0f));
}

// ---------------------------------------------------------------------------
// alpha[b,h,l] = softmax_l(qg . Kh) * L
// ---------------------------------------------------------------------------
__global__ __launch_bounds__(256) void alpha_kernel(
    const float* __restrict__ qg, const float* __restrict__ Kn,
    float* __restrict__ alpha) {
  const int bh = blockIdx.x, b = bh >> 4, h = bh & 15;
  const int tid = threadIdx.x;
  __shared__ float qs[64];
  __shared__ float red[256];
  if (tid < 64) qs[tid] = qg[bh * 64 + tid];
  __syncthreads();
  float sv[16];
  float mx = -1e30f;
#pragma unroll 1
  for (int i = 0; i < 16; ++i) {
    int l = tid + 256 * i;
    const float* kr = Kn + (size_t)(b * L_ + l) * D_ + h * 64;
    float s = 0.f;
#pragma unroll
    for (int d = 0; d < 64; ++d) s += qs[d] * kr[d];
    sv[i] = s;
    mx = fmaxf(mx, s);
  }
  red[tid] = mx; __syncthreads();
  for (int st = 128; st > 0; st >>= 1) {
    if (tid < st) red[tid] = fmaxf(red[tid], red[tid + st]);
    __syncthreads();
  }
  mx = red[0]; __syncthreads();
  float sum = 0.f;
#pragma unroll
  for (int i = 0; i < 16; ++i) { sv[i] = expf(sv[i] - mx); sum += sv[i]; }
  red[tid] = sum; __syncthreads();
  for (int st = 128; st > 0; st >>= 1) {
    if (tid < st) red[tid] += red[tid + st];
    __syncthreads();
  }
  float scl = (float)L_ / red[0];
#pragma unroll
  for (int i = 0; i < 16; ++i)
    alpha[(size_t)bh * L_ + tid + 256 * i] = sv[i] * scl;
}

// ---------------------------------------------------------------------------
// KV[b,h,d,e] = sum_l (phiK[l,d]*alpha[l]) * V[l,e]   — WMMA over K=L.
// ---------------------------------------------------------------------------
__global__ __launch_bounds__(256) void kv_kernel(
    const float* __restrict__ phiK,   // [(b*L+l)*H+h][64]
    const float* __restrict__ alpha,  // [B*H, L]
    const float* __restrict__ Vn,     // [B*L, D]
    float* __restrict__ KV) {         // [B*H, 64, 64]
  const int bh = blockIdx.x, b = bh >> 4, h = bh & 15;
  const int tid = threadIdx.x, lane = tid & 31, w = tid >> 5;
  const int mt = w & 3, nth = w >> 2;
  const int l16 = lane & 15, lh = lane >> 4;
  __shared__ unsigned short At[64 * 32];  // [d][l]
  __shared__ unsigned short Vt[64 * 32];  // [e][l]
  FragAcc acc[2];
#pragma unroll
  for (int ni = 0; ni < 2; ++ni)
#pragma unroll
    for (int r = 0; r < 8; ++r) acc[ni].f[r] = 0.0f;

  for (int l0 = 0; l0 < L_; l0 += 32) {
    int tl = tid & 31, db = (tid >> 5) * 8;
    size_t bl = (size_t)(b * L_ + l0 + tl);
    float al = alpha[(size_t)bh * L_ + l0 + tl];
    const float* pk = phiK + (bl * H_ + h) * 64 + db;
    const float* vv = Vn + bl * D_ + h * 64 + db;
#pragma unroll
    for (int i = 0; i < 8; ++i) {
      At[(db + i) * 32 + tl] = f32_bf16(pk[i] * al);
      Vt[(db + i) * 32 + tl] = f32_bf16(vv[i]);
    }
    __syncthreads();
    FragB16 af;
    int rowA = 16 * mt + l16;
#pragma unroll
    for (int j = 0; j < 8; ++j) {
      int kk = ((j >> 2) << 4) + (lh << 3) + ((j & 3) << 1);
      af.u[j] = *(const unsigned int*)(&At[rowA * 32 + kk]);
    }
#pragma unroll
    for (int ni = 0; ni < 2; ++ni) {
      FragB16 bfr;
      int rowB = 16 * (2 * nth + ni) + l16;
#pragma unroll
      for (int j = 0; j < 8; ++j)
        bfr.u[j] = *(const unsigned int*)(&Vt[rowB * 32 + ((j >> 2) << 4) + (lh << 3) + ((j & 3) << 1)]);
      acc[ni].v = __builtin_amdgcn_wmma_f32_16x16x32_bf16(
          false, af.v, false, bfr.v, (short)0, acc[ni].v, false, false);
    }
    __syncthreads();
  }
#pragma unroll
  for (int ni = 0; ni < 2; ++ni)
#pragma unroll
    for (int r = 0; r < 8; ++r) {
      int d = 16 * mt + 8 * lh + r;
      int e = 16 * (2 * nth + ni) + l16;
      KV[((size_t)bh * 64 + d) * 64 + e] = acc[ni].f[r];
    }
}

// ---------------------------------------------------------------------------
// K_sum[b,h,d] = sum_l phiK[l,d]*alpha[l]  (coalesced: 4 rows in flight)
// ---------------------------------------------------------------------------
__global__ __launch_bounds__(256) void ksum_kernel(
    const float* __restrict__ phiK, const float* __restrict__ alpha,
    float* __restrict__ Ks) {
  const int bh = blockIdx.x, b = bh >> 4, h = bh & 15;
  const int d = threadIdx.x & 63, lg = threadIdx.x >> 6;
  __shared__ float ps[4][64];
  float s = 0.f;
  for (int l = lg; l < L_; l += 4)
    s += phiK[((size_t)(b * L_ + l) * H_ + h) * 64 + d] *
         alpha[(size_t)bh * L_ + l];
  ps[lg][d] = s;
  __syncthreads();
  if (lg == 0) Ks[bh * 64 + d] = ps[0][d] + ps[1][d] + ps[2][d] + ps[3][d];
}

// ---------------------------------------------------------------------------
// den[b,l,h] = phiQ . K_sum + eps
// ---------------------------------------------------------------------------
__global__ __launch_bounds__(256) void den_kernel(
    const float* __restrict__ phiQ, const float* __restrict__ Ks,
    float* __restrict__ den) {
  const int tid = threadIdx.x;
  const int grp = tid >> 6, j = tid & 63;
  const int rid = blockIdx.x * 4 + grp;
  const int bl = rid >> 4, h = rid & 15;
  const int b = bl / L_;
  __shared__ float ps[4][64];
  ps[grp][j] = phiQ[(size_t)rid * 64 + j] * Ks[(b * H_ + h) * 64 + j];
  __syncthreads();
  if (j == 0) {
    float s = 0.f;
    for (int i = 0; i < 64; ++i) s += ps[grp][i];
    den[rid] = s + EPS_ATTN;
  }
}

// ---------------------------------------------------------------------------
// num: out[b,l,h,e] = (phiQ[l,:] @ KV[b,h]) / den.  64 rows (l) per block.
// phiQ tile staged with async-to-LDS when available.
// ---------------------------------------------------------------------------
__global__ __launch_bounds__(256) void num_kernel(
    const unsigned short* __restrict__ phiQb,  // [(b*L+l)*H+h][64] bf16
    const float* __restrict__ KV,              // [B*H, 64, 64]
    const float* __restrict__ den,             // [B*L*H]
    float* __restrict__ Out) {                 // [B*L, D]
  const int blk = blockIdx.x;            // bh * (L/64) + lc
  const int bh = blk >> 6, lc = blk & 63;
  const int b = bh >> 4, h = bh & 15;
  const int l0 = lc * 64;
  const int tid = threadIdx.x, lane = tid & 31, w = tid >> 5;
  const int mt = w & 3, nth = w >> 2;
  const int l16 = lane & 15, lh = lane >> 4;
  __shared__ unsigned short Qs[64 * 64];    // [l][d]
  __shared__ unsigned short KVt[64 * 64];   // [e][d]
#if HAVE_ASYNC
#pragma unroll
  for (int i = 0; i < 2; ++i) {
    int u = tid * 2 + i, row = u >> 3, q = u & 7;
    async_cp16(phiQb + ((size_t)(b * L_ + l0 + row) * H_ + h) * 64 + 8 * q,
               &Qs[row * 64 + 8 * q]);
  }
#pragma unroll 1
  for (int i = 0; i < 16; ++i) {
    int idx = i * 256 + tid;
    int row = idx >> 6, col = idx & 63;
    KVt[col * 64 + row] = f32_bf16(KV[(size_t)bh * 4096 + idx]);
  }
  __builtin_amdgcn_s_wait_asynccnt((short)0);
#else
#pragma unroll 1
  for (int i = 0; i < 16; ++i) {
    int idx = i * 256 + tid;
    int row = idx >> 6, col = idx & 63;
    Qs[row * 64 + col] =
        phiQb[((size_t)(b * L_ + l0 + row) * H_ + h) * 64 + col];
    KVt[col * 64 + row] = f32_bf16(KV[(size_t)bh * 4096 + idx]);
  }
#endif
  __syncthreads();
  FragAcc acc[2];
#pragma unroll
  for (int ni = 0; ni < 2; ++ni)
#pragma unroll
    for (int r = 0; r < 8; ++r) acc[ni].f[r] = 0.0f;
#pragma unroll
  for (int k0 = 0; k0 < 64; k0 += 32) {
    FragB16 af;
    int rowA = 16 * mt + l16;
#pragma unroll
    for (int j = 0; j < 8; ++j) {
      int kk = k0 + ((j >> 2) << 4) + (lh << 3) + ((j & 3) << 1);
      af.u[j] = *(const unsigned int*)(&Qs[rowA * 64 + kk]);
    }
#pragma unroll
    for (int ni = 0; ni < 2; ++ni) {
      FragB16 bfr;
      int rowB = 16 * (2 * nth + ni) + l16;
#pragma unroll
      for (int j = 0; j < 8; ++j)
        bfr.u[j] =
            *(const unsigned int*)(&KVt[rowB * 64 + k0 + ((j >> 2) << 4) + (lh << 3) + ((j & 3) << 1)]);
      acc[ni].v = __builtin_amdgcn_wmma_f32_16x16x32_bf16(
          false, af.v, false, bfr.v, (short)0, acc[ni].v, false, false);
    }
  }
#pragma unroll
  for (int ni = 0; ni < 2; ++ni)
#pragma unroll
    for (int r = 0; r < 8; ++r) {
      int l = 16 * mt + 8 * lh + r;
      int e = 16 * (2 * nth + ni) + l16;
      size_t bl = (size_t)(b * L_ + l0 + l);
      float dn = den[bl * H_ + h];
      Out[bl * D_ + h * 64 + e] = acc[ni].f[r] / dn;
    }
}

// ---------------------------------------------------------------------------
// Host-side launch
// ---------------------------------------------------------------------------
extern "C" void kernel_launch(void* const* d_in, const int* in_sizes, int n_in,
                              void* d_out, int out_size, void* d_ws,
                              size_t ws_size, hipStream_t stream) {
  const float* x     = (const float*)d_in[0];
  const float* WQ    = (const float*)d_in[1];
  const float* bQ    = (const float*)d_in[2];
  const float* gQ    = (const float*)d_in[3];
  const float* betaQ = (const float*)d_in[4];
  const float* WK    = (const float*)d_in[5];
  const float* bK    = (const float*)d_in[6];
  const float* gK    = (const float*)d_in[7];
  const float* betaK = (const float*)d_in[8];
  const float* WV    = (const float*)d_in[9];
  const float* bV    = (const float*)d_in[10];
  const float* gV    = (const float*)d_in[11];
  const float* betaV = (const float*)d_in[12];
  const float* Wg    = (const float*)d_in[13];
  const float* bg    = (const float*)d_in[14];
  const float* gg    = (const float*)d_in[15];
  const float* betag = (const float*)d_in[16];
  const float* cs    = (const float*)d_in[17];
  const float* g_an  = (const float*)d_in[18];
  const float* b_an  = (const float*)d_in[19];
  const float* W1    = (const float*)d_in[20];
  const float* b1    = (const float*)d_in[21];
  const float* W2    = (const float*)d_in[22];
  const float* b2    = (const float*)d_in[23];

  const size_t NT = (size_t)B_ * L_;  // 16384 tokens
  char* ws = (char*)d_ws;
  size_t off = 0;
  auto alloc = [&](size_t bytes) -> void* {
    void* p = ws + off;
    off = (off + bytes + 255) & ~(size_t)255;
    return p;
  };
  unsigned short* xb    = (unsigned short*)alloc(NT * D_ * 2);
  unsigned short* wqT   = (unsigned short*)alloc((size_t)D_ * D_ * 2);
  unsigned short* wkT   = (unsigned short*)alloc((size_t)D_ * D_ * 2);
  unsigned short* wvT   = (unsigned short*)alloc((size_t)D_ * D_ * 2);
  unsigned short* w1T   = (unsigned short*)alloc((size_t)D_ * M_ * 2);
  unsigned short* w2T   = (unsigned short*)alloc((size_t)M_ * D_ * 2);
  float*          raw   = (float*)alloc(NT * D_ * 4);
  float*          Qn    = (float*)alloc(NT * D_ * 4);
  float*          Kn    = (float*)alloc(NT * D_ * 4);
  float*          Vn    = (float*)alloc(NT * D_ * 4);
  float*          phiQf = (float*)alloc(NT * D_ * 4);
  unsigned short* phiQb = (unsigned short*)alloc(NT * D_ * 2);
  float*          phiKf = (float*)alloc(NT * D_ * 4);
  float*          qg    = (float*)alloc((size_t)B_ * H_ * 64 * 4);
  float*          alp   = (float*)alloc((size_t)B_ * H_ * L_ * 4);
  float*          Ks    = (float*)alloc((size_t)B_ * H_ * 64 * 4);
  float*          KV    = (float*)alloc((size_t)B_ * H_ * 64 * 64 * 4);
  float*          den   = (float*)alloc(NT * H_ * 4);
  float*          af    = (float*)alloc(NT * D_ * 4);
  unsigned short* ab    = (unsigned short*)alloc(NT * D_ * 2);
  unsigned short* hb    = (unsigned short*)alloc(NT * M_ * 2);
  float* outp = (float*)d_out;

  // --- bf16 conversions (A operands plain, weights transposed [N][K]) -----
  cvt_bf16_kernel<<<4096, 256, 0, stream>>>(x, xb, (long)NT * D_);
  {
    dim3 gT(D_ / 32, D_ / 32);
    cvt_bf16_T_kernel<<<gT, 256, 0, stream>>>(WQ, wqT, D_, D_);
    cvt_bf16_T_kernel<<<gT, 256, 0, stream>>>(WK, wkT, D_, D_);
    cvt_bf16_T_kernel<<<gT, 256, 0, stream>>>(WV, wvT, D_, D_);
    dim3 g1T(M_ / 32, D_ / 32);   // W1 [D][M] -> [M][D]
    cvt_bf16_T_kernel<<<g1T, 256, 0, stream>>>(W1, w1T, D_, M_);
    dim3 g2T(D_ / 32, M_ / 32);   // W2 [M][D] -> [D][M]
    cvt_bf16_T_kernel<<<g2T, 256, 0, stream>>>(W2, w2T, M_, D_);
  }

  dim3 gQKV(D_ / 128, NT / 128);   // (8, 128)
  dim3 gMLP1(M_ / 128, NT / 128);  // (32, 128)
  dim3 gMLP2(D_ / 128, NT / 128);

  // --- Q/K/V projections + LN ---------------------------------------------
  gemm_bf16_kernel<EPI_NONE><<<gQKV, 256, 0, stream>>>(
      xb, wqT, bQ, raw, nullptr, nullptr, nullptr, (int)NT, D_, D_);
  ln_kernel<<<NT, 256, 0, stream>>>(raw, gQ, betaQ, Qn, nullptr);
  gemm_bf16_kernel<EPI_NONE><<<gQKV, 256, 0, stream>>>(
      xb, wkT, bK, raw, nullptr, nullptr, nullptr, (int)NT, D_, D_);
  ln_kernel<<<NT, 256, 0, stream>>>(raw, gK, betaK, Kn, nullptr);
  gemm_bf16_kernel<EPI_NONE><<<gQKV, 256, 0, stream>>>(
      xb, wvT, bV, raw, nullptr, nullptr, nullptr, (int)NT, D_, D_);
  ln_kernel<<<NT, 256, 0, stream>>>(raw, gV, betaV, Vn, nullptr);

  // --- feature maps --------------------------------------------------------
  phi_kernel<<<NT * H_ / 4, 256, 0, stream>>>(Qn, Wg, bg, gg, betag, cs,
                                              phiQf, phiQb);
  phi_kernel<<<NT * H_ / 4, 256, 0, stream>>>(Kn, Wg, bg, gg, betag, cs,
                                              phiKf, nullptr);

  // --- RALA rescale --------------------------------------------------------
  qglobal_kernel<<<B_ * H_, 256, 0, stream>>>(Qn, qg);
  alpha_kernel<<<B_ * H_, 256, 0, stream>>>(qg, Kn, alp);

  // --- linear attention KV state ------------------------------------------
  kv_kernel<<<B_ * H_, 256, 0, stream>>>(phiKf, alp, Vn, KV);
  ksum_kernel<<<B_ * H_, 256, 0, stream>>>(phiKf, alp, Ks);
  den_kernel<<<NT * H_ / 4, 256, 0, stream>>>(phiQf, Ks, den);
  num_kernel<<<B_ * H_ * (L_ / 64), 256, 0, stream>>>(phiQb, KV, den, raw);

  // --- post-attn norm + MLP + residuals -----------------------------------
  ln_kernel<<<NT, 256, 0, stream>>>(raw, g_an, b_an, af, ab);
  gemm_bf16_kernel<EPI_GELU_BF16><<<gMLP1, 256, 0, stream>>>(
      ab, w1T, b1, nullptr, hb, nullptr, nullptr, (int)NT, M_, D_);
  gemm_bf16_kernel<EPI_RES2><<<gMLP2, 256, 0, stream>>>(
      hb, w2T, b2, outp, nullptr, af, x, (int)NT, D_, M_);
}
